// MultiHeadAttention_9328668967148
// MI455X (gfx1250) — matrix operand order repaired and checked by reading, not compiled, not run on hardware
//
#include <hip/hip_runtime.h>

typedef __attribute__((ext_vector_type(16))) _Float16 v16h;
typedef __attribute__((ext_vector_type(8)))  float    v8f;

#define D_MODEL   768
#define N_HEADS   12
#define HEAD_DIM  64
#define SEQ       2048
#define BATCH     2
#define ROWS      (BATCH * SEQ)        /* 4096 */
#define QKV_COLS  (3 * D_MODEL)        /* 2304 */
/* SCALE * log2(e) = (1/8) * 1.4426950408889634 */
#define QSCALE    0.18033688011112043f

struct __attribute__((aligned(16))) u32x4 { unsigned int x, y, z, w; };
struct U32x8 { u32x4 lo, hi; };

__device__ __forceinline__ v16h make_v16h(u32x4 lo, u32x4 hi) {
    U32x8 u; u.lo = lo; u.hi = hi;
    return __builtin_bit_cast(v16h, u);
}

/* A-operand, 16x32 f16: lane holds row (lane&15); half 0 -> K {0..15},
   half 1 -> K {16..31}. Two b128 loads. */
__device__ __forceinline__ v16h ld_A16x32(const _Float16* base, int ld,
                                          int m0, int kb, int mr, int hf) {
    const _Float16* p = base + (size_t)(m0 + mr) * ld + kb + hf * 16;
    return make_v16h(*(const u32x4*)p, *(const u32x4*)(p + 8));
}

/* B-operand, 32x16 f16: lane holds column; per lane 16 contiguous K halfs
   starting at half*16.  base_k0 points at element [col(lane)][k=0]. */
__device__ __forceinline__ v16h ld_B32x16(const _Float16* base_k0, int hf) {
    const _Float16* p = base_k0 + hf * 16;
    return make_v16h(*(const u32x4*)p, *(const u32x4*)(p + 8));
}

__device__ __forceinline__ v8f wmma_f16(v16h a, v16h b, v8f c) {
    return __builtin_amdgcn_wmma_f32_16x16x32_f16(false, a, false, b,
                                                  (short)0, c, false, false);
}

__device__ __forceinline__ float fast_exp2(float x) {
#if __has_builtin(__builtin_amdgcn_exp2f)
    return __builtin_amdgcn_exp2f(x);
#else
    return exp2f(x);
#endif
}

/* CDNA5 async global->LDS copy of 16 bytes (tracked by ASYNCcnt). */
__device__ __forceinline__ void async_g2l_b128(const _Float16* g, _Float16* l) {
    unsigned           lo = (unsigned)(uintptr_t)l;          /* LDS offset */
    unsigned long long ga = (unsigned long long)(uintptr_t)g;
    asm volatile("global_load_async_to_lds_b128 %0, %1, off"
                 :: "v"(lo), "v"(ga) : "memory");
}
__device__ __forceinline__ void wait_async_le4() {
    asm volatile("s_wait_asynccnt 0x4" ::: "memory");
}

/* ---------------- stage 0: conversions ---------------- */

__global__ void cvt_f32_f16(const float* __restrict__ src,
                            _Float16* __restrict__ dst, int n) {
    int i = blockIdx.x * 256 + threadIdx.x;
    if (i < n) dst[i] = (_Float16)src[i];
}

/* src [rows, cols] row-major -> dst [cols, rows] row-major, f16.
   grid = (ceil(cols/256), rows) */
__global__ void cvtT_f32_f16(const float* __restrict__ src,
                             _Float16* __restrict__ dst, int rows, int cols) {
    int c = blockIdx.x * 256 + threadIdx.x;
    int r = blockIdx.y;
    if (c < cols) dst[(size_t)c * rows + r] = (_Float16)src[(size_t)r * cols + c];
}

/* =======================================================================
   GEMM core: block = 128 threads (4 waves), tile M=128 x N=128, K-step 32.
   B (Wt, [N,K] row-major) staged through double-buffered LDS via
   global_load_async_to_lds_b128, prefetched one K-step ahead.
   Wave w computes rows [m0+w*32, +32) (two A tiles) x all 128 columns:
   acc[2][8]; each LDS B operand feeds two WMMAs.  A global loads are
   register double-buffered across K-steps; B LDS reads use a depth-1
   software prefetch so WMMAs run behind partial dscnt waits.
   ======================================================================= */

#define GEMM_PROLOGUE(LDW)                                                    \
    __shared__ __attribute__((aligned(16))) _Float16 ldsB[2][128 * 32];       \
    const int tid  = threadIdx.x;                                             \
    const int lane = tid & 31;                                                \
    const int wv   = tid >> 5;                                                \
    const int mr   = lane & 15, hf = lane >> 4;                               \
    const int m0   = blockIdx.y * 128 + wv * 32;                              \
    const int n0   = blockIdx.x * 128;                                        \
    v8f zz = {};                                                              \
    v8f acc[2][8] = { { zz, zz, zz, zz, zz, zz, zz, zz },                     \
                      { zz, zz, zz, zz, zz, zz, zz, zz } };                   \
    /* async fill helper: 128 threads x 4 chunks = 128 rows x 32 halfs */     \
    auto fillB = [&](int kb, int buf) {                                       \
        const _Float16* grow = Wt + (size_t)(n0 + tid) * (LDW) + kb;          \
        _Float16* lrow = &ldsB[buf][tid * 32];                                \
        _Pragma("unroll")                                                     \
        for (int c = 0; c < 4; ++c) async_g2l_b128(grow + c * 8, lrow + c * 8); \
    };                                                                        \
    fillB(0, 0);                                                              \
    fillB(32, 1);                                                             \
    wait_async_le4();   /* buf0 resident */                                   \
    __syncthreads();

#define GEMM_KSTEP(APTR, LDA, KDIM)                                           \
    v16h A0c = ld_A16x32((APTR), (LDA), m0,      0, mr, hf);                  \
    v16h A1c = ld_A16x32((APTR), (LDA), m0 + 16, 0, mr, hf);                  \
    int cur = 0;                                                              \
    for (int kb = 0; kb < (KDIM); kb += 32) {                                 \
        /* prefetch next K-step's A tiles (clamped on last iter) */           \
        int ka = kb + 32;                                                     \
        if (ka > (KDIM) - 32) ka = (KDIM) - 32;                               \
        v16h A0n = ld_A16x32((APTR), (LDA), m0,      ka, mr, hf);             \
        v16h A1n = ld_A16x32((APTR), (LDA), m0 + 16, ka, mr, hf);             \
        /* consume LDS B with depth-1 prefetch; each B feeds two WMMAs */     \
        const _Float16* bbase = &ldsB[cur][mr * 32 + hf * 16];                \
        v16h Bc = make_v16h(*(const u32x4*)bbase, *(const u32x4*)(bbase + 8)); \
        _Pragma("unroll")                                                     \
        for (int nt = 0; nt < 8; ++nt) {                                      \
            v16h Bn = Bc;                                                     \
            if (nt < 7) {                                                     \
                const _Float16* bp = bbase + (nt + 1) * (16 * 32);            \
                Bn = make_v16h(*(const u32x4*)bp, *(const u32x4*)(bp + 8));   \
            }                                                                 \
            acc[0][nt] = wmma_f16(A0c, Bc, acc[0][nt]);                       \
            acc[1][nt] = wmma_f16(A1c, Bc, acc[1][nt]);                       \
            Bc = Bn;                                                          \
        }                                                                     \
        __syncthreads();              /* all waves done reading buf(cur) */   \
        int kpre = kb + 64;                                                   \
        if (kpre > (KDIM) - 32) kpre = (KDIM) - 32;                           \
        fillB(kpre, cur);             /* overwrite consumed buffer */         \
        wait_async_le4();             /* buf(cur^1) resident */               \
        __syncthreads();                                                      \
        cur ^= 1;                                                             \
        A0c = A0n;                                                            \
        A1c = A1n;                                                            \
    }

/* ---------------- stage 1: fused QKV projection ----------------
   C[4096,2304] = X @ Wqkv ; scatter to Q,K [B,H,S,64] (Q scaled by
   SCALE*log2e) and V transposed [B,H,64,S]. */
__global__ __launch_bounds__(128)
void gemm_qkv(const _Float16* __restrict__ X, const _Float16* __restrict__ Wt,
              _Float16* __restrict__ Q, _Float16* __restrict__ K,
              _Float16* __restrict__ Vt) {
    GEMM_PROLOGUE(D_MODEL)
    GEMM_KSTEP(X, D_MODEL, D_MODEL)

#pragma unroll
    for (int nt = 0; nt < 8; ++nt) {
        int c   = n0 + nt * 16 + mr;
        int mtx = c / D_MODEL;           /* 0=Q 1=K 2=V */
        int rem = c % D_MODEL;
        int hh  = rem >> 6, d = rem & 63;
#pragma unroll
        for (int mt = 0; mt < 2; ++mt)
#pragma unroll
            for (int r = 0; r < 8; ++r) {
                int g = m0 + mt * 16 + r + 8 * hf;   /* row = b*SEQ + s */
                int b = g >> 11, s = g & (SEQ - 1);
                float v = acc[mt][nt][r];
                size_t headbase = (size_t)(b * N_HEADS + hh);
                if (mtx == 0)
                    Q[(headbase * SEQ + s) * HEAD_DIM + d] = (_Float16)(v * QSCALE);
                else if (mtx == 1)
                    K[(headbase * SEQ + s) * HEAD_DIM + d] = (_Float16)v;
                else
                    Vt[(headbase * HEAD_DIM + d) * SEQ + s] = (_Float16)v;
            }
    }
}

/* ---------------- stage 3: output projection + bias ---------------- */
__global__ __launch_bounds__(128)
void gemm_out(const _Float16* __restrict__ A, const _Float16* __restrict__ Wt,
              const float* __restrict__ bias, float* __restrict__ out) {
    GEMM_PROLOGUE(D_MODEL)
    GEMM_KSTEP(A, D_MODEL, D_MODEL)

#pragma unroll
    for (int nt = 0; nt < 8; ++nt) {
        int c = n0 + nt * 16 + mr;
        float bc = bias[c];
#pragma unroll
        for (int mt = 0; mt < 2; ++mt)
#pragma unroll
            for (int r = 0; r < 8; ++r) {
                int g = m0 + mt * 16 + r + 8 * hf;
                out[(size_t)g * D_MODEL + c] = acc[mt][nt][r] + bc;
            }
    }
}

/* ---------------- stage 2: causal flash attention ----------------
   1 wave = one 16-query tile; 4 independent waves per block (no block
   barriers: per-wave trip counts differ).  K^T operands are register
   double-buffered; V operands are loaded at iteration start so the
   softmax VALU stretch hides their latency. */
__global__ __launch_bounds__(128)
void flash_attn(const _Float16* __restrict__ Q, const _Float16* __restrict__ K,
                const _Float16* __restrict__ Vt, _Float16* __restrict__ O) {
    __shared__ __attribute__((aligned(16))) _Float16 ldsP[4][16 * 48];

    const int lane = threadIdx.x & 31;
    const int wv   = threadIdx.x >> 5;
    const int mr   = lane & 15, hf = lane >> 4;
    const int bh   = blockIdx.y;
    const int q0   = (blockIdx.x * 4 + wv) * 16;

    const _Float16* Qh = Q  + (size_t)bh * SEQ * HEAD_DIM;
    const _Float16* Kh = K  + (size_t)bh * SEQ * HEAD_DIM;
    const _Float16* Vh = Vt + (size_t)bh * HEAD_DIM * SEQ;
    _Float16* lp = ldsP[wv];

    /* Q tile resident in registers (already scaled by SCALE*log2e) */
    v16h Aq0 = ld_A16x32(Qh, HEAD_DIM, q0, 0,  mr, hf);
    v16h Aq1 = ld_A16x32(Qh, HEAD_DIM, q0, 32, mr, hf);

    v8f z = {};
    v8f oa[4] = { z, z, z, z };
    float rm[8], ls[8];
#pragma unroll
    for (int r = 0; r < 8; ++r) { rm[r] = -__builtin_inff(); ls[r] = 0.0f; }

    /* K^T operand loader: tile t (0/1 = keys +0/+16), dh chunk (0/32) */
    auto ldK = [&](int kbase, int t, int dh) {
        return ld_B32x16(Kh + (size_t)(kbase + t * 16 + mr) * HEAD_DIM + dh, hf);
    };

    v16h Bk[4];
    Bk[0] = ldK(0, 0, 0);  Bk[1] = ldK(0, 0, 32);
    Bk[2] = ldK(0, 1, 0);  Bk[3] = ldK(0, 1, 32);

    for (int kb = 0; kb < q0 + 16; kb += 32) {
        /* V tiles for this block: issued early, consumed after softmax */
        v16h Bv[4];
#pragma unroll
        for (int ot = 0; ot < 4; ++ot)
            Bv[ot] = ld_B32x16(Vh + (size_t)(ot * 16 + mr) * SEQ + kb, hf);

        /* scores from resident K operands */
        v8f c0 = {}, c1 = {};
        c0 = wmma_f16(Aq0, Bk[0], c0);
        c0 = wmma_f16(Aq1, Bk[1], c0);
        c1 = wmma_f16(Aq0, Bk[2], c1);
        c1 = wmma_f16(Aq1, Bk[3], c1);

        /* prefetch next block's K^T operands (clamped; unused on last iter) */
        int kn = kb + 32;
        if (kn > SEQ - 32) kn = SEQ - 32;
        v16h Bn[4];
        Bn[0] = ldK(kn, 0, 0);  Bn[1] = ldK(kn, 0, 32);
        Bn[2] = ldK(kn, 1, 0);  Bn[3] = ldK(kn, 1, 32);

        if (kb + 31 > q0 + 15) {            /* uniform tail branch */
#pragma unroll
            for (int r = 0; r < 8; ++r) {
                int qrow = q0 + r + 8 * hf;
                c0[r] = (kb + mr      > qrow) ? -__builtin_inff() : c0[r];
                c1[r] = (kb + 16 + mr > qrow) ? -__builtin_inff() : c1[r];
            }
        }

        /* online softmax (base-2; scale folded into Q) */
#pragma unroll
        for (int r = 0; r < 8; ++r) {
            float nm = fmaxf(c0[r], c1[r]);
            nm = fmaxf(nm, __shfl_xor(nm, 1, 32));
            nm = fmaxf(nm, __shfl_xor(nm, 2, 32));
            nm = fmaxf(nm, __shfl_xor(nm, 4, 32));
            nm = fmaxf(nm, __shfl_xor(nm, 8, 32));
            nm = fmaxf(nm, rm[r]);
            float alpha = fast_exp2(rm[r] - nm);
            rm[r] = nm;
            float p0 = fast_exp2(c0[r] - nm);
            float p1 = fast_exp2(c1[r] - nm);
            float rs = p0 + p1;
            rs += __shfl_xor(rs, 1, 32);
            rs += __shfl_xor(rs, 2, 32);
            rs += __shfl_xor(rs, 4, 32);
            rs += __shfl_xor(rs, 8, 32);
            ls[r] = ls[r] * alpha + rs;
#pragma unroll
            for (int ot = 0; ot < 4; ++ot) oa[ot][r] *= alpha;
            int M = r + 8 * hf;
            lp[M * 48 + mr]      = (_Float16)p0;   /* C-layout -> LDS */
            lp[M * 48 + 16 + mr] = (_Float16)p1;
        }

        asm volatile("s_wait_dscnt 0x0" ::: "memory");

        /* reload P in A-layout (16x32), then PV: O += P @ V */
        const _Float16* pp = lp + mr * 48 + hf * 16;
        v16h Ap = make_v16h(*(const u32x4*)pp, *(const u32x4*)(pp + 8));
#pragma unroll
        for (int ot = 0; ot < 4; ++ot)
            oa[ot] = wmma_f16(Ap, Bv[ot], oa[ot]);

#pragma unroll
        for (int i = 0; i < 4; ++i) Bk[i] = Bn[i];
    }

    /* epilogue: normalize and store head-major O[B,S,H,64] (f16) */
    int b = bh / N_HEADS, hh = bh % N_HEADS;
#pragma unroll
    for (int ot = 0; ot < 4; ++ot)
#pragma unroll
        for (int r = 0; r < 8; ++r) {
            int s = q0 + r + 8 * hf;
            float val = oa[ot][r] / ls[r];
            O[(((size_t)(b * SEQ + s)) * N_HEADS + hh) * HEAD_DIM + ot * 16 + mr] =
                (_Float16)val;
        }
}

/* ---------------- launcher ---------------- */
extern "C" void kernel_launch(void* const* d_in, const int* in_sizes, int n_in,
                              void* d_out, int out_size, void* d_ws, size_t ws_size,
                              hipStream_t stream) {
    const float* x      = (const float*)d_in[0];
    /* d_in[1] = attn_mask (causal, known statically -> unused) */
    const float* W_qkv  = (const float*)d_in[2];
    const float* W_proj = (const float*)d_in[3];
    const float* b_proj = (const float*)d_in[4];

    char*  ws  = (char*)d_ws;
    size_t off = 0;
    auto alloc = [&](size_t bytes) -> void* {
        void* p = ws + off;
        off += (bytes + 255) & ~(size_t)255;
        return p;
    };

    const size_t headElems = (size_t)BATCH * N_HEADS * SEQ * HEAD_DIM;
    _Float16* xb  = (_Float16*)alloc((size_t)ROWS * D_MODEL * 2);
    _Float16* wqt = (_Float16*)alloc((size_t)QKV_COLS * D_MODEL * 2);
    _Float16* wpt = (_Float16*)alloc((size_t)D_MODEL * D_MODEL * 2);
    _Float16* Qb  = (_Float16*)alloc(headElems * 2);
    _Float16* Kb  = (_Float16*)alloc(headElems * 2);
    _Float16* Vtb = (_Float16*)alloc(headElems * 2);
    _Float16* Ob  = (_Float16*)alloc((size_t)ROWS * D_MODEL * 2);
    (void)ws_size; (void)in_sizes; (void)n_in; (void)out_size;

    int nx = ROWS * D_MODEL;
    cvt_f32_f16<<<(nx + 255) / 256, 256, 0, stream>>>(x, xb, nx);
    cvtT_f32_f16<<<dim3((QKV_COLS + 255) / 256, D_MODEL), 256, 0, stream>>>(
        W_qkv, wqt, D_MODEL, QKV_COLS);
    cvtT_f32_f16<<<dim3((D_MODEL + 255) / 256, D_MODEL), 256, 0, stream>>>(
        W_proj, wpt, D_MODEL, D_MODEL);

    gemm_qkv<<<dim3(QKV_COLS / 128, ROWS / 128), 128, 0, stream>>>(xb, wqt, Qb, Kb, Vtb);
    flash_attn<<<dim3(SEQ / 64, BATCH * N_HEADS), 128, 0, stream>>>(Qb, Kb, Vtb, Ob);
    gemm_out<<<dim3(D_MODEL / 128, ROWS / 128), 128, 0, stream>>>(Ob, wpt, b_proj,
                                                                  (float*)d_out);
}
